// TemporalSNN_72885595013560
// MI455X (gfx1250) — matrix operand order repaired and checked by reading, not compiled, not run on hardware
//
#include <hip/hip_runtime.h>
#include <cstdint>
#include <cstddef>

// ---------------------------------------------------------------------------
// Temporal SNN on gfx1250 (MI455X), layer-sequential formulation.
//
//   spikes_in[t] = (rand < sigmoid(x))            (binary, exact in f16)
//   per layer:  syn = a*syn + S @ W + b ; mem = B*mem + syn ; spk = mem>1 ; mem -= spk
//
// 4 kernels: spikegen -> layer0 -> layer1 -> layer2(+acc).
// GEMMs use v_wmma_f32_16x16x32_f16; weights pre-swizzled into LDS in the
// B-operand lanes, contiguous k order (k = 16*half + elem); LIF state lives in registers in
// the C/D layout; inner k-loop is software-pipelined with depth-2 register
// double-buffering in consume-then-refill order (no operand copies: each
// slot is read by its WMMA and refilled by loads landing 2 chunks ahead).
// ---------------------------------------------------------------------------

typedef _Float16 v16h __attribute__((ext_vector_type(16)));
typedef _Float16 v8h  __attribute__((ext_vector_type(8)));
typedef _Float16 v4h  __attribute__((ext_vector_type(4)));
typedef float    v8f  __attribute__((ext_vector_type(8)));

#define TS      100
#define BATCH   256
#define DIM_IN  2048
#define DIM_H   1024
#define DIM_OUT 256

#define SNN_ALPHA 0.8187307530779818f   // exp(-1/5)
#define SNN_BETA  0.9048374180359595f   // exp(-1/10)

// ---------------------------------------------------------------------------
// Kernel 1: input spike generation.
//   rnd layout [B][T][IN] (f32)  ->  spk layout [T][B][IN] (f16, {0,1})
// ---------------------------------------------------------------------------
__global__ void snn_spikegen(const float* __restrict__ x,
                             const float* __restrict__ rnd,
                             _Float16* __restrict__ spk)
{
    const int I4 = DIM_IN / 4;
    int64_t gid = (int64_t)blockIdx.x * blockDim.x + threadIdx.x;
    int i4 = (int)(gid % I4);
    int t  = (int)((gid / I4) % TS);
    int b  = (int)(gid / ((int64_t)I4 * TS));
    if (b >= BATCH) return;

    float4 r  = ((const float4*)rnd)[((int64_t)b * TS + t) * I4 + i4];
    float4 xv = ((const float4*)x)[(int64_t)b * I4 + i4];

    v4h s;
    s[0] = (_Float16)((r.x < 1.0f / (1.0f + __expf(-xv.x))) ? 1.0f : 0.0f);
    s[1] = (_Float16)((r.y < 1.0f / (1.0f + __expf(-xv.y))) ? 1.0f : 0.0f);
    s[2] = (_Float16)((r.z < 1.0f / (1.0f + __expf(-xv.z))) ? 1.0f : 0.0f);
    s[3] = (_Float16)((r.w < 1.0f / (1.0f + __expf(-xv.w))) ? 1.0f : 0.0f);

    ((v4h*)spk)[((int64_t)t * BATCH + b) * I4 + i4] = s;
}

// ---------------------------------------------------------------------------
// Layer kernel. WG = 256 threads = 8 waves = 4 m-tile-waves x 2 n-tile-waves;
// each wave owns 1 m-tile (16 rows) x NPW n-tiles (16 cols each), so the WG
// output tile is 64 x (32*NPW) for ALL timesteps.  W slab [K x NT] sits in
// LDS (f16), pre-swizzled to a B-fragment layout with contiguous k:
//   frag(kc,nt): lane = 16*((k>>4)&1) + (n&15), elem = k&15   (k in chunk)
// A-fragment (16-bit A 16x32): lane m=lane&15, half=lane>>4:
//   elems 0..7  = K[kc*32 + half*16 + 0..7]
//   elems 8..15 = K[kc*32 + half*16 + 8..15]      (two 16B global loads)
// C/D layout: VGPR r -> row (r + 8*half), col = lane&15 -> LIF state mapping.
// One A-fragment feeds NPW WMMAs (register reuse halves A traffic for NPW=2).
// ---------------------------------------------------------------------------
template <int K, int NPW, bool LAST>
__global__ void snn_layer(const _Float16* __restrict__ spkIn,   // [T][B][K]
                          const float*    __restrict__ W,       // [K][N]
                          const float*    __restrict__ bias,    // [N]
                          _Float16*       __restrict__ spkOut,  // [T][B][N] (!LAST)
                          float*          __restrict__ accOut,  // [B][N]    (LAST)
                          int N)
{
    extern __shared__ _Float16 smemH[];          // K*NT halves, fragment order
    constexpr int NT = 32 * NPW;                 // neuron cols per WG
    constexpr int FR = NT / 16;                  // frags per k-chunk
    constexpr int KC = K / 32;                   // k-chunks

    const int tid   = threadIdx.x;
    const int nBase = blockIdx.x * NT;
    const int bBase = blockIdx.y * 64;

    // ---- Phase 1: stage + swizzle the weight slab into LDS (f16) ----------
    for (int idx = tid; idx < K * NT; idx += 256) {
        int k = idx / NT;                        // row of W
        int n = idx % NT;                        // col within slab (coalesced)
        float w = W[(int64_t)k * N + nBase + n];
        int frag = (k >> 5) * FR + (n >> 4);
        int lane = (((k >> 4) & 1) << 4) | (n & 15);
        int e    = k & 15;
        smemH[((frag << 5) + lane) * 16 + e] = (_Float16)w;
    }
    __syncthreads();

    const int wave   = tid >> 5;
    const int lane   = tid & 31;
    const int half   = lane >> 4;
    const int lcol   = lane & 15;
    const int mtile  = wave & 3;                 // 0..3 (16 rows each)
    const int ntile0 = (wave >> 2) * NPW;        // first n-tile of this wave

    // A stream base: row = bBase + mtile*16 + (lane&15), K-offset half*16
    const _Float16* aBase =
        spkIn + (int64_t)(bBase + mtile * 16 + lcol) * K + half * 16;

    float bn[NPW];
    #pragma unroll
    for (int j = 0; j < NPW; ++j)
        bn[j] = bias[nBase + (ntile0 + j) * 16 + lcol];

    float syn[NPW][8] = {}, mem[NPW][8] = {}, acc[NPW][8] = {};

    const int stateRow = bBase + mtile * 16 + half * 8;    // + r
    int stateCol[NPW];
    #pragma unroll
    for (int j = 0; j < NPW; ++j)
        stateCol[j] = nBase + (ntile0 + j) * 16 + lcol;

    for (int t = 0; t < TS; ++t) {
        const _Float16* aT = aBase + (int64_t)t * BATCH * K;
        if (t + 1 < TS)                          // overlap whole GEMM
            __builtin_prefetch(aT + (int64_t)BATCH * K, 0, 1);

        // ---- depth-2 software-pipelined k-loop ----
        v8h  alo[2], ahi[2];
        v16h bfr[2][NPW];
        auto stage = [&](int kc, int s) {
            alo[s] = *(const v8h*)(aT + kc * 32);
            ahi[s] = *(const v8h*)(aT + kc * 32 + 8);
            #pragma unroll
            for (int j = 0; j < NPW; ++j)
                bfr[s][j] = *(const v16h*)(
                    smemH + (((kc * FR + ntile0 + j) << 5) + lane) * 16);
        };
        stage(0, 0);
        stage(1, 1);

        v8f c[NPW];
        #pragma unroll
        for (int j = 0; j < NPW; ++j) c[j] = (v8f){};

        // consume-then-refill: WMMA reads slot s at issue, then the refill
        // loads for chunk kc+2 are issued into the same slot -> no copies,
        // operands always fetched 2 chunks ahead of their consumer.
        #pragma unroll 2
        for (int kc = 0; kc < KC - 2; ++kc) {
            const int s = kc & 1;
            v16h a = __builtin_shufflevector(alo[s], ahi[s],
                        0,1,2,3,4,5,6,7,8,9,10,11,12,13,14,15);
            #pragma unroll
            for (int j = 0; j < NPW; ++j)
                c[j] = __builtin_amdgcn_wmma_f32_16x16x32_f16(
                           false, a, false, bfr[s][j], (short)0, c[j], false, false);
            stage(kc + 2, s);
        }
        #pragma unroll
        for (int kc = KC - 2; kc < KC; ++kc) {   // pipeline drain
            const int s = kc & 1;
            v16h a = __builtin_shufflevector(alo[s], ahi[s],
                        0,1,2,3,4,5,6,7,8,9,10,11,12,13,14,15);
            #pragma unroll
            for (int j = 0; j < NPW; ++j)
                c[j] = __builtin_amdgcn_wmma_f32_16x16x32_f16(
                           false, a, false, bfr[s][j], (short)0, c[j], false, false);
        }

        // ---- LIF state update in the C/D register layout ----
        #pragma unroll
        for (int j = 0; j < NPW; ++j) {
            #pragma unroll
            for (int r = 0; r < 8; ++r) {
                float s = SNN_ALPHA * syn[j][r] + c[j][r] + bn[j];
                float m = SNN_BETA  * mem[j][r] + s;
                float sp = (m > 1.0f) ? 1.0f : 0.0f;
                m -= sp;
                syn[j][r] = s;
                mem[j][r] = m;
                if (LAST) {
                    acc[j][r] += sp;
                } else {
                    spkOut[((int64_t)t * BATCH + (stateRow + r)) * N +
                           stateCol[j]] = (_Float16)sp;
                }
            }
        }
    }

    if (LAST) {
        #pragma unroll
        for (int j = 0; j < NPW; ++j)
            #pragma unroll
            for (int r = 0; r < 8; ++r)
                accOut[(int64_t)(stateRow + r) * N + stateCol[j]] = acc[j][r];
    }
}

// ---------------------------------------------------------------------------
// Host-side launcher.  d_ws layout (150 MB needed):
//   [0, 100MB)        spk_in  f16 [T][B][2048]   (freed after layer0, reused)
//   [100MB, 150MB)    spk0    f16 [T][B][1024]
//   [0, 50MB)         spk1    f16 [T][B][1024]   (overlaps dead spk_in)
// ---------------------------------------------------------------------------
extern "C" void kernel_launch(void* const* d_in, const int* in_sizes, int n_in,
                              void* d_out, int out_size, void* d_ws, size_t ws_size,
                              hipStream_t stream)
{
    (void)in_sizes; (void)n_in; (void)out_size; (void)ws_size;

    const float* x   = (const float*)d_in[0];
    const float* rnd = (const float*)d_in[1];
    const float* W0  = (const float*)d_in[2];
    const float* b0  = (const float*)d_in[3];
    const float* W1  = (const float*)d_in[4];
    const float* b1  = (const float*)d_in[5];
    const float* W2  = (const float*)d_in[6];
    const float* b2  = (const float*)d_in[7];
    float* out = (float*)d_out;

    char* ws = (char*)d_ws;
    const size_t spkInBytes = (size_t)TS * BATCH * DIM_IN * sizeof(_Float16); // 100 MB
    _Float16* spkIn = (_Float16*)ws;
    _Float16* spk0  = (_Float16*)(ws + spkInBytes);
    _Float16* spk1  = (_Float16*)ws;      // reuse spk_in region (dead after layer0)

    // Dynamic LDS: layer0 slab is 256 KB (<= 320 KB/WGP), layer1 128 KB.
    (void)hipFuncSetAttribute(reinterpret_cast<const void*>(&snn_layer<DIM_IN, 2, false>),
                              hipFuncAttributeMaxDynamicSharedMemorySize, DIM_IN * 64 * 2);
    (void)hipFuncSetAttribute(reinterpret_cast<const void*>(&snn_layer<DIM_H, 2, false>),
                              hipFuncAttributeMaxDynamicSharedMemorySize, DIM_H * 64 * 2);
    (void)hipFuncSetAttribute(reinterpret_cast<const void*>(&snn_layer<DIM_H, 1, true>),
                              hipFuncAttributeMaxDynamicSharedMemorySize, DIM_H * 32 * 2);

    // 1) input spikes: one thread per 4 elements
    {
        int64_t total4 = (int64_t)BATCH * TS * (DIM_IN / 4);
        int blocks = (int)((total4 + 255) / 256);
        snn_spikegen<<<blocks, 256, 0, stream>>>(x, rnd, spkIn);
    }
    // 2) layer 0: [T] x (256x2048 @ 2048x1024) ; 16x4 = 64 WGs, 256 KB LDS
    snn_layer<DIM_IN, 2, false>
        <<<dim3(DIM_H / 64, BATCH / 64), 256, DIM_IN * 64 * 2, stream>>>
        (spkIn, W0, b0, spk0, nullptr, DIM_H);
    // 3) layer 1: [T] x (256x1024 @ 1024x1024) ; 64 WGs, 128 KB LDS
    snn_layer<DIM_H, 2, false>
        <<<dim3(DIM_H / 64, BATCH / 64), 256, DIM_H * 64 * 2, stream>>>
        (spk0, W1, b1, spk1, nullptr, DIM_H);
    // 4) layer 2 + spike accumulation -> d_out (256x256 f32) ; 8x4 = 32 WGs
    snn_layer<DIM_H, 1, true>
        <<<dim3(DIM_OUT / 32, BATCH / 64), 256, DIM_H * 32 * 2, stream>>>
        (spk1, W2, b2, nullptr, out, DIM_OUT);
}
